// TransformerSeqLayer_12833362280492
// MI455X (gfx1250) — matrix operand order repaired and checked by reading, not compiled, not run on hardware
//
#include <hip/hip_runtime.h>
#include <hip/hip_bf16.h>

// ---------------------------------------------------------------------------
// MI455X (gfx1250) transformer layer: WMMA-bf16 GEMMs + flash attention.
// Wave32, WGP. All matmuls go through v_wmma_f32_16x16x32_bf16.
// GEMM K-loop is double-buffered: global loads for tile k+1 are in flight
// (register-staged) while tile k's WMMAs execute; one barrier per iteration.
// ---------------------------------------------------------------------------

typedef __attribute__((ext_vector_type(16))) __bf16 v16bf;
typedef __attribute__((ext_vector_type(8)))  __bf16 v8bf;
typedef __attribute__((ext_vector_type(4)))  __bf16 v4bf;
typedef __attribute__((ext_vector_type(8)))  float  v8f;
typedef __attribute__((ext_vector_type(4)))  float  v4f;

__device__ __forceinline__ v8f wmma_bf16(v16bf a, v16bf b, v8f c) {
  // 8 args: (neg_a, A, neg_b, B, c_mod, C, reuse_a, reuse_b)
  return __builtin_amdgcn_wmma_f32_16x16x32_bf16(false, a, false, b,
                                                 (short)0, c, false, false);
}

// Build a 16x32 bf16 A-fragment from two contiguous 8-element LDS chunks.
// ISA layout: element e -> K = (e&7) + 16*(e>>3) + 8*(lane>>4)
__device__ __forceinline__ v16bf frag_from_chunks(const __bf16* p0, const __bf16* p1) {
  v8bf lo = *(const v8bf*)p0;
  v8bf hi = *(const v8bf*)p1;
  v16bf r;
#pragma unroll
  for (int i = 0; i < 8; i++) { r[i] = lo[i]; r[i + 8] = hi[i]; }
  return r;
}

// Register staging type for the A tile: raw global data (convert at store).
template <typename AT> struct ARegT { v4f v[8]; };
template <> struct ARegT<__bf16> { v4bf v[8]; };

// ---------------------------------------------------------------------------
// GEMM: C[MR x N] = act( A[MR x K] @ B[K x N] + bias )
// A: fp32 or bf16 row-major (converted to bf16 in LDS); B: fp32 row-major.
// Block tile 128x128, K-tile 64, double-buffered LDS. 256 threads = 8 waves;
// wave = 32x64 = 2x4 WMMA tiles. Requires MR%128==0, N%128==0, K%64==0.
// ---------------------------------------------------------------------------
template <typename AT, bool RELU, bool OUT_BF16>
__global__ __launch_bounds__(256) void gemm_wmma_kernel(
    const AT* __restrict__ A, const float* __restrict__ B,
    const float* __restrict__ bias, void* __restrict__ Cout,
    int MR, int N, int K) {
  __shared__ alignas(32) __bf16 As[2][128][64];   // [buf][m][k]
  __shared__ alignas(32) __bf16 Bs[2][128][64];   // [buf][n][k] (transposed)

  const int t = threadIdx.x;
  const int lane = t & 31, w = t >> 5;
  const int half = lane >> 4, l16 = lane & 15;
  const int mblock = blockIdx.y * 128, nblock = blockIdx.x * 128;
  const int mwave = (w >> 1) * 32, nwave = (w & 1) * 64;

  // staging thread->tile mapping
  const int acol = (t & 15) * 4;   // A: 16 col-chunks of 4
  const int arow0 = t >> 4;        // A: rows arow0 + 16*i
  const int bnn = (t & 31) * 4;    // B: 32 n-chunks of 4
  const int bkr = t >> 5;          // B: k rows bkr + 8*i

  ARegT<AT> ar;
  v4f br[8];

  auto loadA = [&](int k0) {
#pragma unroll
    for (int i = 0; i < 8; i++) {
      const size_t gi = (size_t)(mblock + arow0 + i * 16) * K + k0 + acol;
      if constexpr (sizeof(AT) == 2)
        ar.v[i] = *(const v4bf*)(A + gi);
      else
        ar.v[i] = *(const v4f*)(A + gi);
    }
  };
  auto loadB = [&](int k0) {
#pragma unroll
    for (int i = 0; i < 8; i++) {
      const size_t gi = (size_t)(k0 + bkr + i * 8) * N + nblock + bnn;
      br[i] = *(const v4f*)(B + gi);
    }
  };
  auto storeTiles = [&](int bufi) {
#pragma unroll
    for (int i = 0; i < 8; i++) {
      v4bf o4;
      if constexpr (sizeof(AT) == 2) {
        o4 = ar.v[i];
      } else {
#pragma unroll
        for (int j = 0; j < 4; j++) o4[j] = (__bf16)ar.v[i][j];
      }
      *(v4bf*)&As[bufi][arow0 + i * 16][acol] = o4;
    }
#pragma unroll
    for (int i = 0; i < 8; i++) {
#pragma unroll
      for (int j = 0; j < 4; j++)
        Bs[bufi][bnn + j][bkr + i * 8] = (__bf16)br[i][j];
    }
  };

  v8f acc[2][4];
#pragma unroll
  for (int i = 0; i < 2; i++)
#pragma unroll
    for (int j = 0; j < 4; j++)
#pragma unroll
      for (int r = 0; r < 8; r++) acc[i][j][r] = 0.0f;

  // prologue: stage tile 0
  loadA(0);
  loadB(0);
  storeTiles(0);
  __syncthreads();

  const int nk = K / 64;
  for (int kt = 0; kt < nk; kt++) {
    const int cur = kt & 1;
    const bool more = (kt + 1 < nk);
    if (more) {  // issue next tile's global loads; they fly under the WMMAs
      loadA((kt + 1) * 64);
      loadB((kt + 1) * 64);
    }

#pragma unroll
    for (int ks = 0; ks < 2; ks++) {
      const int kb = ks * 32;
      v16bf af[2];
#pragma unroll
      for (int mt = 0; mt < 2; mt++) {
        const __bf16* rowp = &As[cur][mwave + mt * 16 + l16][0];
        af[mt] = frag_from_chunks(rowp + kb + half * 8,
                                  rowp + kb + 16 + half * 8);
      }
#pragma unroll
      for (int nt = 0; nt < 4; nt++) {
        const __bf16* rowp = &Bs[cur][nwave + nt * 16 + l16][0];
        // B layout: element e -> K = (e&7) + 16*(e>>3) + 8*(lane>>4), same as A
        v16bf bfr = frag_from_chunks(rowp + kb + half * 8, rowp + kb + 16 + half * 8);
#pragma unroll
        for (int mt = 0; mt < 2; mt++)
          acc[mt][nt] = wmma_bf16(af[mt], bfr, acc[mt][nt]);
      }
    }

    if (more) {
      storeTiles(cur ^ 1);  // other buffer: last read finished before the
      __syncthreads();      // barrier that ended iteration kt-1
    }
  }

  // --- epilogue: D rows r + 8*half, cols lane&15 ---
#pragma unroll
  for (int mt = 0; mt < 2; mt++) {
#pragma unroll
    for (int nt = 0; nt < 4; nt++) {
      const int col = nblock + nwave + nt * 16 + l16;
      const float bv = bias ? bias[col] : 0.0f;
#pragma unroll
      for (int r = 0; r < 8; r++) {
        const int row = mblock + mwave + mt * 16 + r + half * 8;
        float vv = acc[mt][nt][r] + bv;
        if (RELU) vv = fmaxf(vv, 0.0f);
        const size_t gi = (size_t)row * N + col;
        if constexpr (OUT_BF16)
          ((__bf16*)Cout)[gi] = (__bf16)vv;
        else
          ((float*)Cout)[gi] = vv;
      }
    }
  }
}

// ---------------------------------------------------------------------------
// Flash attention: one block = 128 query rows of one (b,head).
// 8 waves x 16 rows. Iterate key tiles of 128; online softmax; WMMA for
// S = Q K^T and ctx += P V. scale = 1/sqrt(64) = 0.125.
// q/k/v stored as [B*M, 1024] with head slice at col head*64.
// ---------------------------------------------------------------------------
__global__ __launch_bounds__(256) void flash_attn_kernel(
    const float* __restrict__ Q, const float* __restrict__ Kt,
    const float* __restrict__ V, float* __restrict__ Ctx) {
  constexpr int Mseq = 1024, Dm = 1024;
  __shared__ alignas(32) __bf16 kbuf[128][64];     // [key][hd]
  __shared__ alignas(32) __bf16 vbuf[64][128];     // [hd][key] (transposed)
  __shared__ alignas(32) __bf16 pbuf[8][16][128];  // per-wave P tile

  const int t = threadIdx.x, lane = t & 31, w = t >> 5;
  const int half = lane >> 4, l16 = lane & 15;
  const int q0 = blockIdx.x * 128;
  const int bh = blockIdx.y;
  const int b = bh >> 4, head = bh & 15;
  const size_t baseRow = (size_t)b * Mseq;

  // Preload Q A-fragments for this wave's 16 rows (hd = 64 -> 2 frags)
  v16bf qa[2];
  {
    const int qrow = q0 + w * 16 + l16;
    const float* qp = Q + (baseRow + qrow) * Dm + head * 64;
#pragma unroll
    for (int f = 0; f < 2; f++) {
      v16bf a;
#pragma unroll
      for (int c = 0; c < 2; c++) {
        const int off = f * 32 + c * 16 + half * 8;
        v4f x0 = *(const v4f*)(qp + off);
        v4f x1 = *(const v4f*)(qp + off + 4);
#pragma unroll
        for (int j = 0; j < 4; j++) {
          a[c * 8 + j] = (__bf16)x0[j];
          a[c * 8 + 4 + j] = (__bf16)x1[j];
        }
      }
      qa[f] = a;
    }
  }

  float mrow[8], lrow[8];
  v8f ctxa[4];
#pragma unroll
  for (int r = 0; r < 8; r++) { mrow[r] = -3.0e38f; lrow[r] = 0.0f; }
#pragma unroll
  for (int vt = 0; vt < 4; vt++)
#pragma unroll
    for (int r = 0; r < 8; r++) ctxa[vt][r] = 0.0f;

  for (int kt = 0; kt < 8; kt++) {
    // cooperative K/V staging (V transposed) to LDS, fp32 -> bf16
    {
      const int cc = (t & 7) * 8;
#pragma unroll
      for (int i = 0; i < 4; i++) {
        const int kr = (t >> 3) + i * 32;
        const size_t grow = (baseRow + kt * 128 + kr) * Dm + head * 64 + cc;
        v4f k0v = *(const v4f*)(Kt + grow);
        v4f k1v = *(const v4f*)(Kt + grow + 4);
        v8bf k8;
#pragma unroll
        for (int j = 0; j < 4; j++) { k8[j] = (__bf16)k0v[j]; k8[4 + j] = (__bf16)k1v[j]; }
        *(v8bf*)&kbuf[kr][cc] = k8;
        v4f v0v = *(const v4f*)(V + grow);
        v4f v1v = *(const v4f*)(V + grow + 4);
#pragma unroll
        for (int j = 0; j < 4; j++) {
          vbuf[cc + j][kr] = (__bf16)v0v[j];
          vbuf[cc + 4 + j][kr] = (__bf16)v1v[j];
        }
      }
    }
    __syncthreads();

    // pull next K/V tile toward the WGP while this tile computes
    if (kt + 1 < 8) {
      const size_t nrow = (baseRow + (kt + 1) * 128 + (t >> 1)) * Dm +
                          head * 64 + (t & 1) * 32;
      __builtin_prefetch(Kt + nrow, 0, 1);  // -> global_prefetch_b8
      __builtin_prefetch(V + nrow, 0, 1);
    }

    // S = Q K^T : 16 rows x 128 keys, contraction over hd=64 (2 WMMA k-steps)
    v8f s[8];
#pragma unroll
    for (int nt = 0; nt < 8; nt++) {
#pragma unroll
      for (int r = 0; r < 8; r++) s[nt][r] = 0.0f;
      const __bf16* rowp = &kbuf[nt * 16 + l16][0];
#pragma unroll
      for (int ks = 0; ks < 2; ks++) {
        v16bf bfr = frag_from_chunks(rowp + ks * 32 + half * 8, rowp + ks * 32 + 16 + half * 8);
        s[nt] = wmma_bf16(qa[ks], bfr, s[nt]);
      }
    }

    const float scale = 0.125f;  // 1/sqrt(64)
    // online softmax (rows live per 16-lane half; reduce via xor 1,2,4,8)
#pragma unroll
    for (int r = 0; r < 8; r++) {
      float mx = -3.0e38f;
#pragma unroll
      for (int nt = 0; nt < 8; nt++) mx = fmaxf(mx, s[nt][r]);
      mx *= scale;
      mx = fmaxf(mx, __shfl_xor(mx, 1, 32));
      mx = fmaxf(mx, __shfl_xor(mx, 2, 32));
      mx = fmaxf(mx, __shfl_xor(mx, 4, 32));
      mx = fmaxf(mx, __shfl_xor(mx, 8, 32));
      const float mnew = fmaxf(mrow[r], mx);
      const float alpha = __expf(mrow[r] - mnew);
      mrow[r] = mnew;
      float psum = 0.0f;
#pragma unroll
      for (int nt = 0; nt < 8; nt++) {
        const float p = __expf(s[nt][r] * scale - mnew);
        psum += p;
        pbuf[w][r + half * 8][nt * 16 + l16] = (__bf16)p;
      }
      psum += __shfl_xor(psum, 1, 32);
      psum += __shfl_xor(psum, 2, 32);
      psum += __shfl_xor(psum, 4, 32);
      psum += __shfl_xor(psum, 8, 32);
      lrow[r] = lrow[r] * alpha + psum;
#pragma unroll
      for (int vt = 0; vt < 4; vt++) ctxa[vt][r] *= alpha;
    }

    // ctx += P @ V : contraction over 128 keys (4 WMMA k-steps), hd N=64
#pragma unroll
    for (int ks2 = 0; ks2 < 4; ks2++) {
      const __bf16* prow = &pbuf[w][l16][0];
      v16bf pa = frag_from_chunks(prow + ks2 * 32 + half * 8,
                                  prow + ks2 * 32 + 16 + half * 8);
#pragma unroll
      for (int vt = 0; vt < 4; vt++) {
        const __bf16* vrow = &vbuf[vt * 16 + l16][0];
        v16bf bfr = frag_from_chunks(vrow + ks2 * 32 + half * 8, vrow + ks2 * 32 + 16 + half * 8);
        ctxa[vt] = wmma_bf16(pa, bfr, ctxa[vt]);
      }
    }
    __syncthreads();
  }

  // normalize and write ctx (fp32)
#pragma unroll
  for (int vt = 0; vt < 4; vt++) {
    const int col = head * 64 + vt * 16 + l16;
#pragma unroll
    for (int r = 0; r < 8; r++) {
      const int row = q0 + w * 16 + r + half * 8;
      const float inv = 1.0f / lrow[r];
      Ctx[(baseRow + row) * Dm + col] = ctxa[vt][r] * inv;
    }
  }
}

// ---------------------------------------------------------------------------
// out = LayerNorm(X + Y) * gamma + beta, row width 1024, block per row.
// ---------------------------------------------------------------------------
__global__ __launch_bounds__(256) void add_ln_kernel(
    const float* __restrict__ X, const float* __restrict__ Y,
    const float* __restrict__ G, const float* __restrict__ Bt,
    float* __restrict__ Out) {
  const int row = blockIdx.x, t = threadIdx.x;
  const float* xr = X + (size_t)row * 1024;
  const float* yr = Y + (size_t)row * 1024;
  float v[4];
  float s = 0.f, s2 = 0.f;
#pragma unroll
  for (int i = 0; i < 4; i++) {
    const int c = t + i * 256;
    const float vv = xr[c] + yr[c];
    v[i] = vv; s += vv; s2 += vv * vv;
  }
#pragma unroll
  for (int m = 16; m >= 1; m >>= 1) {
    s += __shfl_xor(s, m, 32);
    s2 += __shfl_xor(s2, m, 32);
  }
  __shared__ float ss[8], ss2[8];
  if ((t & 31) == 0) { ss[t >> 5] = s; ss2[t >> 5] = s2; }
  __syncthreads();
  s = 0.f; s2 = 0.f;
#pragma unroll
  for (int i = 0; i < 8; i++) { s += ss[i]; s2 += ss2[i]; }
  const float mu = s * (1.0f / 1024.0f);
  const float var = s2 * (1.0f / 1024.0f) - mu * mu;
  const float rs = rsqrtf(var + 1e-5f);
#pragma unroll
  for (int i = 0; i < 4; i++) {
    const int c = t + i * 256;
    Out[(size_t)row * 1024 + c] = (v[i] - mu) * rs * G[c] + Bt[c];
  }
}

// ---------------------------------------------------------------------------
extern "C" void kernel_launch(void* const* d_in, const int* in_sizes, int n_in,
                              void* d_out, int out_size, void* d_ws, size_t ws_size,
                              hipStream_t stream) {
  const float* h     = (const float*)d_in[0];
  const float* h_all = (const float*)d_in[1];
  const float* Wq    = (const float*)d_in[2];
  const float* Wk    = (const float*)d_in[3];
  const float* Wv    = (const float*)d_in[4];
  const float* Wo    = (const float*)d_in[5];
  const float* W1    = (const float*)d_in[6];
  const float* b1    = (const float*)d_in[7];
  const float* W2    = (const float*)d_in[8];
  const float* b2    = (const float*)d_in[9];
  const float* g1    = (const float*)d_in[10];
  const float* be1   = (const float*)d_in[11];
  const float* g2    = (const float*)d_in[12];
  const float* be2   = (const float*)d_in[13];
  float* out = (float*)d_out;

  const int Bb = 8, Mm = 1024, Dd = 1024, Ff = 4096;
  const int BM = Bb * Mm;  // 8192 rows

  // Workspace: [q][k][v][ctx] fp32 (32 MB each) + ffmid bf16 (64 MB) = 192 MB
  char* ws = (char*)d_ws;
  const size_t SZ = (size_t)BM * Dd * sizeof(float);
  float* qb   = (float*)(ws);
  float* kb   = (float*)(ws + SZ);
  float* vb   = (float*)(ws + 2 * SZ);
  float* ctxb = (float*)(ws + 3 * SZ);
  __bf16* ffmid = (__bf16*)(ws + 4 * SZ);
  float* attn_out = qb;  // q dead after attention
  float* h1       = kb;  // k dead after attention
  float* ff       = vb;  // v dead after attention

  dim3 blk(256);
  // q/k/v projections
  gemm_wmma_kernel<float, false, false><<<dim3(Dd / 128, BM / 128), blk, 0, stream>>>(
      h, Wq, nullptr, qb, BM, Dd, Dd);
  gemm_wmma_kernel<float, false, false><<<dim3(Dd / 128, BM / 128), blk, 0, stream>>>(
      h_all, Wk, nullptr, kb, BM, Dd, Dd);
  gemm_wmma_kernel<float, false, false><<<dim3(Dd / 128, BM / 128), blk, 0, stream>>>(
      h_all, Wv, nullptr, vb, BM, Dd, Dd);
  // attention
  flash_attn_kernel<<<dim3(Mm / 128, Bb * 16), blk, 0, stream>>>(qb, kb, vb, ctxb);
  // output projection
  gemm_wmma_kernel<float, false, false><<<dim3(Dd / 128, BM / 128), blk, 0, stream>>>(
      ctxb, Wo, nullptr, attn_out, BM, Dd, Dd);
  // h1 = LN(h + attn_out)
  add_ln_kernel<<<dim3(BM), blk, 0, stream>>>(h, attn_out, g1, be1, h1);
  // ffmid = relu(h1 @ W1 + b1) in bf16
  gemm_wmma_kernel<float, true, true><<<dim3(Ff / 128, BM / 128), blk, 0, stream>>>(
      h1, W1, b1, (void*)ffmid, BM, Ff, Dd);
  // ff = ffmid @ W2 + b2
  gemm_wmma_kernel<__bf16, false, false><<<dim3(Dd / 128, BM / 128), blk, 0, stream>>>(
      ffmid, W2, b2, ff, BM, Dd, Ff);
  // out = LN(h1 + ff)
  add_ln_kernel<<<dim3(BM), blk, 0, stream>>>(h1, ff, g2, be2, out);
}
